// BysMamba_46420006535991
// MI455X (gfx1250) — matrix operand order repaired and checked by reading, not compiled, not run on hardware
//
#include <hip/hip_runtime.h>
#include <hip/hip_bf16.h>

// ---------------------------------------------------------------------------
// BysMamba forward for MI455X (gfx1250, wave32, WMMA).
// Dense math: v_wmma_f32_16x16x32_bf16 (bf16 in, f32 acc), LDS-staged tiles
// laid out to match the CDNA5 WMMA fragment layouts so A fragment builds are
// contiguous ds_load_b128s. Sequential selective scan keeps state in VGPRs
// with double-buffered LDS broadcast of the shared B/C vectors.
// ---------------------------------------------------------------------------

#define DIMSZ 768
#define DSTATE 16
#define DCONV 4
#define DI 1536
#define DTRANK 48
#define NVOCAB 768
#define NBLK 10
#define LSEQ 2048

typedef __attribute__((ext_vector_type(16))) __bf16 v16bf;
typedef __attribute__((ext_vector_type(4)))  __bf16 v4bf;
typedef __attribute__((ext_vector_type(8)))  float v8f;

__device__ __forceinline__ float silu_f(float v) {
  return v * (1.0f / (1.0f + __expf(-v)));
}
__device__ __forceinline__ float softplus_f(float v) {
  return (v > 30.0f) ? v : log1pf(__expf(v));
}

// ---------------------------------------------------------------------------
// Embedding + 4x4/stride-4 patch "conv" (VOCAB==DIM so output adds directly).
// ---------------------------------------------------------------------------
__global__ __launch_bounds__(256) void embed_patch_kernel(
    const int* __restrict__ x, const float* __restrict__ emb,
    const float* __restrict__ pw, const float* __restrict__ pb,
    float* __restrict__ h)
{
  const int l = blockIdx.x;           // 0..2047
  __shared__ float pv[16];
  __shared__ int tok;
  if (threadIdx.x < 16) {
    const int hh = l >> 9;            // l / 512
    const int ww = l & 511;           // l % 512
    const int i = threadIdx.x >> 2, j = threadIdx.x & 3;
    pv[threadIdx.x] = (float)x[(4 * hh + i) * 2048 + 4 * ww + j];
    if (threadIdx.x == 0) tok = x[8 * 2048 + l];  // x[:, 0, 8, :]
  }
  __syncthreads();
  const int t = tok;
  for (int d = threadIdx.x; d < DIMSZ; d += 256) {
    float acc = emb[(size_t)t * DIMSZ + d] + pb[d];
#pragma unroll
    for (int k = 0; k < 16; ++k) acc += pv[k] * pw[d * 16 + k];
    h[(size_t)l * DIMSZ + d] = acc;
  }
}

// ---------------------------------------------------------------------------
// Tiled WMMA GEMM:  C[M x N] (+)= act( A[M x K] * W[N x K]^T + bias )
// 256 threads = 8 waves (2x4); block tile 64x128; wave tile 32x32 (2x2 WMMA).
//  As: row-major [m][k], stride 48 elems (96B, b128-aligned) — A frag = 2xb128
//  Bs: K-major  [k][n], stride 136 elems (272B, b128-aligned) — B frag (lane
//      = N col, 16 K values) = 272B-strided reads per lane.
// act: 0 = none, 1 = silu, 2 = softplus.
// ---------------------------------------------------------------------------
#define TM 64
#define TN 128
#define TK 32
#define APAD 16
#define BPAD 8

__global__ __launch_bounds__(256) void wmma_gemm_kernel(
    const float* __restrict__ A, int lda,
    const float* __restrict__ W, int ldw,
    const float* __restrict__ bias,
    float* __restrict__ C,
    int M, int N, int K, int act, int accumulate)
{
  __shared__ __bf16 As[TM][TK + APAD];
  __shared__ __bf16 Bs[TK][TN + BPAD];

  const int tid  = threadIdx.x;
  const int lane = tid & 31;
  const int wave = tid >> 5;
  const int wm   = wave >> 2;          // 0..1
  const int wn   = wave & 3;           // 0..3
  const int blockM = blockIdx.y * TM;
  const int blockN = blockIdx.x * TN;
  const bool fullM = (blockM + TM <= M);
  const bool fullN = (blockN + TN <= N);

  v8f acc[2][2] = {};

  for (int k0 = 0; k0 < K; k0 += TK) {
    const bool fullK = (k0 + TK <= K);

    // ---- stage A tile (f32 -> bf16) ----
    if (fullM && fullK) {
      // branch-free: float4 coalesced loads, cvt_pk, b64 LDS stores
#pragma unroll
      for (int t = 0; t < 2; ++t) {
        const int u = tid + t * 256;           // 0..511
        const int r = u >> 3;                  // TK/4 = 8 units per row
        const int c = (u & 7) << 2;
        const float4 f =
            *(const float4*)&A[(size_t)(blockM + r) * lda + k0 + c];
        v4bf b;
        b[0] = (__bf16)f.x; b[1] = (__bf16)f.y;
        b[2] = (__bf16)f.z; b[3] = (__bf16)f.w;
        *(v4bf*)&As[r][c] = b;
      }
    } else {
      for (int i = tid; i < TM * TK; i += 256) {
        const int r = i >> 5, c = i & 31;
        const int gm = blockM + r, gk = k0 + c;
        float v = (gm < M && gk < K) ? A[(size_t)gm * lda + gk] : 0.0f;
        As[r][c] = (__bf16)v;
      }
    }

    // ---- stage B tile (f32 -> bf16, transposed to K-major) ----
    if (fullN && fullK) {
#pragma unroll
      for (int t = 0; t < 4; ++t) {
        const int u = tid + t * 256;           // 0..1023
        const int k = u & 31;                  // contiguous k across lanes
        const int n = ((u >> 5) & 31) << 2;    // 4 consecutive n per unit
        const float* wp = &W[(size_t)(blockN + n) * ldw + k0 + k];
        v4bf b;
        b[0] = (__bf16)wp[0];
        b[1] = (__bf16)wp[(size_t)ldw];
        b[2] = (__bf16)wp[2 * (size_t)ldw];
        b[3] = (__bf16)wp[3 * (size_t)ldw];
        *(v4bf*)&Bs[k][n] = b;
      }
    } else {
      for (int i = tid; i < TK * TN; i += 256) {
        const int k = i >> 7, n = i & 127;     // TN = 128
        const int gn = blockN + n, gk = k0 + k;
        float v = (gn < N && gk < K) ? W[(size_t)gn * ldw + gk] : 0.0f;
        Bs[k][n] = (__bf16)v;
      }
    }

    // prefetch next K tile toward the WGP while WMMAs drain
    if (k0 + TK < K) {
      if (fullM)
        __builtin_prefetch(&A[(size_t)(blockM + (tid & 63)) * lda + k0 + TK],
                           0, 1);
      if (fullN)
        __builtin_prefetch(&W[(size_t)(blockN + (tid & 127)) * ldw + k0 + TK],
                           0, 1);
    }
    __syncthreads();

    // ---- fragments (CDNA5 16-bit layouts) ----
    // A(16x32): lane<16 row=lane kbase 0; lane>=16 kbase 8; elems 0..7 from
    //           kbase, elems 8..15 from 16+kbase -> two contiguous 16B runs.
    // B(32x16): col=lane&15; elems follow the same K order as A.
    const int row0 = wm * 32, col0 = wn * 32;
    const int r  = lane & 15;
    const int kb = (lane >> 4) << 3;
    v16bf afrag[2], bfrag[2];
#pragma unroll
    for (int t = 0; t < 2; ++t) {
#pragma unroll
      for (int e = 0; e < 8; ++e) {
        afrag[t][e]     = As[row0 + t * 16 + r][kb + e];
        afrag[t][e + 8] = As[row0 + t * 16 + r][16 + kb + e];
      }
#pragma unroll
      for (int e = 0; e < 8; ++e) {
        bfrag[t][e]     = Bs[kb + e][col0 + t * 16 + r];
        bfrag[t][e + 8] = Bs[16 + kb + e][col0 + t * 16 + r];
      }
    }
#pragma unroll
    for (int i = 0; i < 2; ++i)
#pragma unroll
      for (int j2 = 0; j2 < 2; ++j2)
        acc[i][j2] = __builtin_amdgcn_wmma_f32_16x16x32_bf16(
            false, afrag[i], false, bfrag[j2],
            (short)0, acc[i][j2], false, false);
    __syncthreads();
  }

  // ---- epilogue: C/D layout — VGPR e: lanes 0-15 M=e, lanes 16-31 M=e+8 ----
#pragma unroll
  for (int i = 0; i < 2; ++i) {
#pragma unroll
    for (int j2 = 0; j2 < 2; ++j2) {
#pragma unroll
      for (int e = 0; e < 8; ++e) {
        const int m = blockM + wm * 32 + i * 16 + e + ((lane >> 4) << 3);
        const int n = blockN + wn * 32 + j2 * 16 + (lane & 15);
        if (m < M && n < N) {
          float v = acc[i][j2][e];
          if (bias) v += bias[n];
          if (act == 1) v = silu_f(v);
          else if (act == 2) v = softplus_f(v);
          const size_t idx = (size_t)m * N + n;
          if (accumulate) C[idx] += v;
          else C[idx] = v;
        }
      }
    }
  }
}

// ---------------------------------------------------------------------------
// Depthwise causal conv (k=4) over the sequence + bias + SiLU.
// dir=+1: y[l] = sum_j xz[l-3+j]*w[j];  dir=-1: y[l] = sum_j xz[l+3-j]*w[j].
// ---------------------------------------------------------------------------
__global__ __launch_bounds__(256) void conv_silu_kernel(
    const float* __restrict__ xz, const float* __restrict__ cw,
    const float* __restrict__ cb, float* __restrict__ xc, int dir)
{
  const int idx = blockIdx.x * 256 + threadIdx.x;
  if (idx >= LSEQ * DI) return;
  const int l = idx / DI;
  const int d = idx - l * DI;
  float acc = cb[d];
#pragma unroll
  for (int j = 0; j < DCONV; ++j) {
    const int pos = l + dir * (j - (DCONV - 1));
    if (pos >= 0 && pos < LSEQ)
      acc += xz[(size_t)pos * (2 * DI) + d] * cw[d * DCONV + j];
  }
  xc[(size_t)l * DI + d] = silu_f(acc);
}

// ---------------------------------------------------------------------------
// Selective scan. 6 blocks x 256 threads == 1536 channels, one channel per
// thread, 16-state in VGPRs. Per-step B/C (shared across all channels) are
// double-buffered through LDS -> one barrier per sequential step.
// Fuses the D skip-connection and silu(z) gating.
// ---------------------------------------------------------------------------
__global__ __launch_bounds__(256) void scan_kernel(
    const float* __restrict__ dt, const float* __restrict__ dbc,
    const float* __restrict__ xc, const float* __restrict__ xz,
    const float* __restrict__ A_log, const float* __restrict__ Dp,
    float* __restrict__ y, int dir)
{
  const int d = blockIdx.x * 256 + threadIdx.x;   // 0..1535
  float A[DSTATE], hs[DSTATE];
#pragma unroll
  for (int s = 0; s < DSTATE; ++s) {
    A[s]  = -__expf(A_log[(size_t)d * DSTATE + s]);
    hs[s] = 0.0f;
  }
  const float Dd = Dp[d];
  __shared__ float Bsh[2][DSTATE], Csh[2][DSTATE];

  // preload step 0
  {
    const int l0 = (dir > 0) ? 0 : (LSEQ - 1);
    if (threadIdx.x < 32) {
      const int s = threadIdx.x & 15;
      const float v = dbc[(size_t)l0 * (DTRANK + 2 * DSTATE) + DTRANK +
                          (threadIdx.x < 16 ? s : DSTATE + s)];
      if (threadIdx.x < 16) Bsh[0][s] = v; else Csh[0][s] = v;
    }
  }
  int p = 0;
  for (int t = 0; t < LSEQ; ++t) {
    const int l = (dir > 0) ? t : (LSEQ - 1 - t);
    __syncthreads();                      // buf p ready for everyone
    if (threadIdx.x < 32 && t + 1 < LSEQ) {
      const int ln = (dir > 0) ? (t + 1) : (LSEQ - 2 - t);
      const int s = threadIdx.x & 15;
      const float v = dbc[(size_t)ln * (DTRANK + 2 * DSTATE) + DTRANK +
                          (threadIdx.x < 16 ? s : DSTATE + s)];
      if (threadIdx.x < 16) Bsh[p ^ 1][s] = v; else Csh[p ^ 1][s] = v;
    }
    const float dtl = dt[(size_t)l * DI + d];
    const float xl  = xc[(size_t)l * DI + d];
    const float zz  = xz[(size_t)l * (2 * DI) + DI + d];
    const float dx  = dtl * xl;
    float accv = 0.0f;
#pragma unroll
    for (int s = 0; s < DSTATE; ++s) {
      hs[s] = __expf(dtl * A[s]) * hs[s] + dx * Bsh[p][s];
      accv += hs[s] * Csh[p][s];
    }
    float yy = accv + xl * Dd;
    yy *= silu_f(zz);
    y[(size_t)l * DI + d] = yy;
    p ^= 1;
  }
}

// ---------------------------------------------------------------------------
// Host-side orchestration
// ---------------------------------------------------------------------------
static void launch_gemm(const float* A, int lda, const float* W, int ldw,
                        const float* bias, float* C, int M, int N, int K,
                        int act, int accumulate, hipStream_t stream)
{
  dim3 grid((N + TN - 1) / TN, (M + TM - 1) / TM);
  wmma_gemm_kernel<<<grid, 256, 0, stream>>>(A, lda, W, ldw, bias, C,
                                             M, N, K, act, accumulate);
}

static void run_block_pass(const float* hin, float* h,
                           const float* in_proj_i, const float* conv_w_i,
                           const float* conv_b_i, const float* x_proj_i,
                           const float* dt_w_i, const float* dt_b_i,
                           const float* A_log_i, const float* D_i,
                           const float* out_proj_i,
                           float* xz, float* xc, float* dbc, float* dt,
                           float* y, int dir, hipStream_t stream)
{
  // xz = hin @ in_proj^T        (2048 x 3072, K=768)
  launch_gemm(hin, DIMSZ, in_proj_i, DIMSZ, nullptr, xz,
              LSEQ, 2 * DI, DIMSZ, 0, 0, stream);
  // xc = silu(causal_conv(xz[:, :DI]) + cb)
  conv_silu_kernel<<<(LSEQ * DI + 255) / 256, 256, 0, stream>>>(
      xz, conv_w_i, conv_b_i, xc, dir);
  // dbc = xc @ x_proj^T         (2048 x 80, K=1536)
  launch_gemm(xc, DI, x_proj_i, DI, nullptr, dbc,
              LSEQ, DTRANK + 2 * DSTATE, DI, 0, 0, stream);
  // dt = softplus(dbc[:, :48] @ dt_w^T + dt_b)   (2048 x 1536, K=48)
  launch_gemm(dbc, DTRANK + 2 * DSTATE, dt_w_i, DTRANK, dt_b_i, dt,
              LSEQ, DI, DTRANK, 2, 0, stream);
  // selective scan + D skip + silu(z) gate -> y
  scan_kernel<<<DI / 256, 256, 0, stream>>>(dt, dbc, xc, xz, A_log_i, D_i,
                                            y, dir);
  // h += y @ out_proj^T         (2048 x 768, K=1536)
  launch_gemm(y, DI, out_proj_i, DI, nullptr, h,
              LSEQ, DIMSZ, DI, 0, 1, stream);
}

extern "C" void kernel_launch(void* const* d_in, const int* in_sizes, int n_in,
                              void* d_out, int out_size, void* d_ws, size_t ws_size,
                              hipStream_t stream)
{
  const int*   x        = (const int*)  d_in[0];
  const float* emb      = (const float*)d_in[1];
  const float* patch_w  = (const float*)d_in[2];
  const float* patch_b  = (const float*)d_in[3];
  const float* in_proj  = (const float*)d_in[4];
  const float* conv_w   = (const float*)d_in[5];
  const float* conv_b   = (const float*)d_in[6];
  const float* x_proj   = (const float*)d_in[7];
  const float* dt_w     = (const float*)d_in[8];
  const float* dt_b     = (const float*)d_in[9];
  const float* A_log    = (const float*)d_in[10];
  const float* Dp       = (const float*)d_in[11];
  const float* out_proj = (const float*)d_in[12];
  const float* lm_head  = (const float*)d_in[13];
  float* out = (float*)d_out;

  // workspace carve-up (floats)
  float* ws  = (float*)d_ws;
  float* h   = ws;                         // 2048*768
  float* hin = h   + (size_t)LSEQ * DIMSZ; // 2048*768
  float* xz  = hin + (size_t)LSEQ * DIMSZ; // 2048*3072
  float* xc  = xz  + (size_t)LSEQ * 2 * DI;// 2048*1536
  float* dbc = xc  + (size_t)LSEQ * DI;    // 2048*80
  float* dt  = dbc + (size_t)LSEQ * (DTRANK + 2 * DSTATE); // 2048*1536
  float* y   = dt  + (size_t)LSEQ * DI;    // 2048*1536

  // h = embed + patch conv
  embed_patch_kernel<<<LSEQ, 256, 0, stream>>>(x, emb, patch_w, patch_b, h);

  for (int i = 0; i < NBLK; ++i) {
    const float* ip = in_proj  + (size_t)i * 2 * DI * DIMSZ;
    const float* cw = conv_w   + (size_t)i * DI * DCONV;
    const float* cb = conv_b   + (size_t)i * DI;
    const float* xp = x_proj   + (size_t)i * (DTRANK + 2 * DSTATE) * DI;
    const float* dw = dt_w     + (size_t)i * DI * DTRANK;
    const float* db = dt_b     + (size_t)i * DI;
    const float* al = A_log    + (size_t)i * DI * DSTATE;
    const float* dd = Dp       + (size_t)i * DI;
    const float* op = out_proj + (size_t)i * DIMSZ * DI;

    // snapshot residual input so both directions of a bidir block read it
    hipMemcpyAsync(hin, h, (size_t)LSEQ * DIMSZ * sizeof(float),
                   hipMemcpyDeviceToDevice, stream);
    run_block_pass(hin, h, ip, cw, cb, xp, dw, db, al, dd, op,
                   xz, xc, dbc, dt, y, +1, stream);
    if (i == 0 || i == NBLK - 1)  // bidirectional first and last block
      run_block_pass(hin, h, ip, cw, cb, xp, dw, db, al, dd, op,
                     xz, xc, dbc, dt, y, -1, stream);
  }

  // logits = h @ lm_head^T  (2048 x 768, K=768)
  launch_gemm(h, DIMSZ, lm_head, DIMSZ, nullptr, out,
              LSEQ, NVOCAB, DIMSZ, 0, 0, stream);
}
